// TernBinLayer_20306605376070
// MI455X (gfx1250) — matrix operand order repaired and checked by reading, not compiled, not run on hardware
//
#include <hip/hip_runtime.h>
#include <hip/hip_bf16.h>

// Ternary-weight GEMM on gfx1250: Y[M,N] = X[M,K](f32) @ W[K,N](ternary int).
// Fast path (needs ~168MB workspace):
//   pass 1: X -> Xhi/Xlo (bf16 two-term split; W is exact in bf16)
//   pass 2: W -> Wt = W^T as bf16 (LDS-tiled transpose)
//   pass 3: double-buffered LDS GEMM (BK=64: 2 WMMA K-sub-steps per stage),
//           v_wmma_f32_16x16x32_bf16, GLOBAL_LOAD_ASYNC_TO_LDS_B128 staging
//           (ASYNCcnt-tracked), one barrier per 64-wide K-step.
// Fallback (small ws): single-buffer kernel with in-loop conversion.

typedef __attribute__((ext_vector_type(16))) __bf16 v16bf_t;
typedef __attribute__((ext_vector_type(8)))  __bf16 v8bf_t;
typedef __attribute__((ext_vector_type(4)))  __bf16 v4bf_t;
typedef __attribute__((ext_vector_type(8)))  float  v8f_t;

#define DIN  4096
#define DOUT 4096
#define BM   128
#define BN   128
#define BK   64     // staged K per barrier (2 WMMA sub-steps of 32)

// ---------------------------------------------------------------------------
// CDNA5 async global->LDS copy (16 bytes per lane), tracked by ASYNCcnt.
// ---------------------------------------------------------------------------
__device__ __forceinline__ void async_copy_b128(const void* g, void* lds) {
    const unsigned lds_addr = (unsigned)(unsigned long long)(uintptr_t)lds;
    const unsigned long long gaddr = (unsigned long long)(uintptr_t)g;
    asm volatile("global_load_async_to_lds_b128 %0, %1, off"
                 :: "v"(lds_addr), "v"(gaddr)
                 : "memory");
}

__device__ __forceinline__ void wait_async0() {
#if __has_builtin(__builtin_amdgcn_s_wait_asynccnt)
    __builtin_amdgcn_s_wait_asynccnt(0);
#else
    asm volatile("s_wait_asynccnt 0x0" ::: "memory");
#endif
}

// ---------------------------------------------------------------------------
// Pass 1: X (f32) -> Xhi, Xlo (bf16). 4 elements per thread.
// ---------------------------------------------------------------------------
__global__ __launch_bounds__(256)
void cvt_x_kernel(const float* __restrict__ X,
                  unsigned short* __restrict__ Xhi,
                  unsigned short* __restrict__ Xlo) {
    const size_t base = ((size_t)blockIdx.x * 256 + threadIdx.x) * 4;
    const float4 v = *(const float4*)(X + base);
    const float f[4] = { v.x, v.y, v.z, v.w };
    v4bf_t h, l;
    #pragma unroll
    for (int j = 0; j < 4; ++j) {
        const __bf16 hj = (__bf16)f[j];
        h[j] = hj;
        l[j] = (__bf16)(f[j] - (float)hj);
    }
    *(v4bf_t*)(Xhi + base) = h;
    *(v4bf_t*)(Xlo + base) = l;
}

// ---------------------------------------------------------------------------
// Pass 2: W[K][N] (int) -> Wt[N][K] (bf16), 64x64 LDS-tiled transpose.
// ---------------------------------------------------------------------------
__global__ __launch_bounds__(256)
void cvt_w_kernel(const int* __restrict__ W, unsigned short* __restrict__ Wt) {
    __shared__ __bf16 t[64][64 + 8];
    const int tid = threadIdx.x;
    const int k0 = blockIdx.y * 64;
    const int n0 = blockIdx.x * 64;
    #pragma unroll
    for (int i = 0; i < 4; ++i) {
        const int idx = tid + i * 256;
        const int kr  = idx >> 4;
        const int c4  = (idx & 15) * 4;
        const int4 w = *(const int4*)(W + (size_t)(k0 + kr) * DOUT + n0 + c4);
        t[c4 + 0][kr] = (__bf16)(float)w.x;
        t[c4 + 1][kr] = (__bf16)(float)w.y;
        t[c4 + 2][kr] = (__bf16)(float)w.z;
        t[c4 + 3][kr] = (__bf16)(float)w.w;
    }
    __syncthreads();
    #pragma unroll
    for (int i = 0; i < 2; ++i) {
        const int idx = tid + i * 256;
        const int nr  = idx >> 3;
        const int sg  = (idx & 7) * 8;
        *(v8bf_t*)(Wt + (size_t)(n0 + nr) * DIN + k0 + sg) = *(const v8bf_t*)&t[nr][sg];
    }
}

// ---------------------------------------------------------------------------
// Pass 3: main GEMM. Double-buffered LDS (96KB), async-to-LDS staging,
// bf16 hi/lo WMMA with f32 accumulate. 32 WMMAs per wave per barrier.
// ---------------------------------------------------------------------------
__global__ __launch_bounds__(256)
void tern_gemm_wmma_pre(const unsigned short* __restrict__ Xhi,
                        const unsigned short* __restrict__ Xlo,
                        const unsigned short* __restrict__ Wt,
                        float* __restrict__ Y) {
    __shared__ __bf16 sAhi[2][BM][BK];   // 32 KB
    __shared__ __bf16 sAlo[2][BM][BK];   // 32 KB
    __shared__ __bf16 sB[2][BN][BK];     // 32 KB, [n][k]

    const int tid   = threadIdx.x;
    const int lane  = tid & 31;
    const int wave  = tid >> 5;
    const int waveM = wave & 3;
    const int waveN = wave >> 2;
    const int half  = lane >> 4;
    const int l16   = lane & 15;

    const int bm = blockIdx.y * BM;
    const int bn = blockIdx.x * BN;

    // Staging: tile is 128 rows x 64 bf16 = 1024 16B chunks; 4 chunks/thread.
    // chunk c (= tid + i*256): row = c>>3, seg = (c&7)*8.
    const int rS = tid >> 3;                  // 0..31
    const int sS = (tid & 7) * 8;             // 0..56

    v8f_t acc[2][4];
    #pragma unroll
    for (int mf = 0; mf < 2; ++mf)
        #pragma unroll
        for (int nf = 0; nf < 4; ++nf) {
            v8f_t z = {};
            acc[mf][nf] = z;
        }

    // Issue all 12 async copies of one 64-wide K tile into buffer `buf`.
    auto stage_tile = [&](int buf, int kt) {
        #pragma unroll
        for (int i = 0; i < 4; ++i) {
            const int row = rS + i * 32;                       // 0..127
            const size_t gA = (size_t)(bm + row) * DIN + kt + sS;
            const size_t gB = (size_t)(bn + row) * DIN + kt + sS;
            async_copy_b128(Xhi + gA, &sAhi[buf][row][sS]);
            async_copy_b128(Xlo + gA, &sAlo[buf][row][sS]);
            async_copy_b128(Wt  + gB, &sB[buf][row][sS]);
        }
    };

    // Prologue: stage tile 0 into buffer 0.
    stage_tile(0, 0);
    wait_async0();
    __syncthreads();

    const int NT = DIN / BK;   // 64
    for (int t = 0; t < NT; ++t) {
        const int cur = t & 1;
        const bool more = (t + 1 < NT);

        // Next tile's async copies complete under the 32 WMMAs below.
        if (more) stage_tile(cur ^ 1, (t + 1) * BK);

        #pragma unroll
        for (int ks = 0; ks < 2; ++ks) {           // two 32-wide sub-steps
            const int kb = ks * 32;
            // Fragments from current buffer (ISA VGPR layouts).
            v16bf_t afh[2], afl[2], bfr[4];
            #pragma unroll
            for (int mf = 0; mf < 2; ++mf) {
                const int row = waveM * 32 + mf * 16 + l16;
                union { v16bf_t v; v8bf_t h[2]; } u;
                const __bf16* ph = &sAhi[cur][row][kb + half * 8];
                u.h[0] = *(const v8bf_t*)(ph);
                u.h[1] = *(const v8bf_t*)(ph + 16);
                afh[mf] = u.v;
                const __bf16* pl = &sAlo[cur][row][kb + half * 8];
                u.h[0] = *(const v8bf_t*)(pl);
                u.h[1] = *(const v8bf_t*)(pl + 16);
                afl[mf] = u.v;
            }
            #pragma unroll
            for (int nf = 0; nf < 4; ++nf) {
                const int col = waveN * 64 + nf * 16 + l16;
                union { v16bf_t v; v8bf_t h[2]; } u;
                const __bf16* pb = &sB[cur][col][kb + half * 8];
                u.h[0] = *(const v8bf_t*)(pb);
                u.h[1] = *(const v8bf_t*)(pb + 16);
                bfr[nf] = u.v;
            }
            #pragma unroll
            for (int mf = 0; mf < 2; ++mf)
                #pragma unroll
                for (int nf = 0; nf < 4; ++nf) {
                    acc[mf][nf] = __builtin_amdgcn_wmma_f32_16x16x32_bf16(
                        false, afh[mf], false, bfr[nf], (short)0, acc[mf][nf], false, false);
                    acc[mf][nf] = __builtin_amdgcn_wmma_f32_16x16x32_bf16(
                        false, afl[mf], false, bfr[nf], (short)0, acc[mf][nf], false, false);
                }
        }

        wait_async0();
        __syncthreads();
    }

    #pragma unroll
    for (int mf = 0; mf < 2; ++mf) {
        #pragma unroll
        for (int nf = 0; nf < 4; ++nf) {
            const int gn = bn + waveN * 64 + nf * 16 + l16;
            #pragma unroll
            for (int r = 0; r < 8; ++r) {
                const int gm = bm + waveM * 32 + mf * 16 + half * 8 + r;
                Y[(size_t)gm * DOUT + gn] = acc[mf][nf][r];
            }
        }
    }
}

// ---------------------------------------------------------------------------
// Fallback: single-buffer kernel with in-loop conversion (no workspace).
// ---------------------------------------------------------------------------
__global__ __launch_bounds__(256)
void tern_gemm_wmma_fallback(const float* __restrict__ X,
                             const int*   __restrict__ W,
                             float*       __restrict__ Y) {
    __shared__ __bf16 sAhi[BM][32];
    __shared__ __bf16 sAlo[BM][32];
    __shared__ __bf16 sB[BN][32];

    const int tid   = threadIdx.x;
    const int lane  = tid & 31;
    const int wave  = tid >> 5;
    const int waveM = wave & 3;
    const int waveN = wave >> 2;
    const int half  = lane >> 4;
    const int l16   = lane & 15;
    const int bm = blockIdx.y * BM;
    const int bn = blockIdx.x * BN;

    v8f_t acc[2][4];
    #pragma unroll
    for (int mf = 0; mf < 2; ++mf)
        #pragma unroll
        for (int nf = 0; nf < 4; ++nf) { v8f_t z = {}; acc[mf][nf] = z; }

    for (int kt = 0; kt < DIN; kt += 32) {
        #pragma unroll
        for (int i = 0; i < 4; ++i) {
            const int idx = tid + i * 256;
            const int row = idx >> 3;
            const int c4  = (idx & 7) * 4;
            const float4 v = *(const float4*)(X + (size_t)(bm + row) * DIN + kt + c4);
            const float f[4] = { v.x, v.y, v.z, v.w };
            #pragma unroll
            for (int j = 0; j < 4; ++j) {
                const __bf16 h = (__bf16)f[j];
                sAhi[row][c4 + j] = h;
                sAlo[row][c4 + j] = (__bf16)(f[j] - (float)h);
            }
        }
        #pragma unroll
        for (int i = 0; i < 4; ++i) {
            const int idx = tid + i * 256;
            const int kr  = idx >> 5;
            const int c4  = (idx & 31) * 4;
            const int4 w = *(const int4*)(W + (size_t)(kt + kr) * DOUT + bn + c4);
            sB[c4 + 0][kr] = (__bf16)(float)w.x;
            sB[c4 + 1][kr] = (__bf16)(float)w.y;
            sB[c4 + 2][kr] = (__bf16)(float)w.z;
            sB[c4 + 3][kr] = (__bf16)(float)w.w;
        }
        __syncthreads();

        v16bf_t afh[2], afl[2], bfr[4];
        #pragma unroll
        for (int mf = 0; mf < 2; ++mf) {
            const int row = waveM * 32 + mf * 16 + l16;
            union { v16bf_t v; v8bf_t h[2]; } u;
            const __bf16* ph = &sAhi[row][half * 8];
            u.h[0] = *(const v8bf_t*)(ph); u.h[1] = *(const v8bf_t*)(ph + 16);
            afh[mf] = u.v;
            const __bf16* pl = &sAlo[row][half * 8];
            u.h[0] = *(const v8bf_t*)(pl); u.h[1] = *(const v8bf_t*)(pl + 16);
            afl[mf] = u.v;
        }
        #pragma unroll
        for (int nf = 0; nf < 4; ++nf) {
            const int col = waveN * 64 + nf * 16 + l16;
            union { v16bf_t v; v8bf_t h[2]; } u;
            const __bf16* pb = &sB[col][half * 8];
            u.h[0] = *(const v8bf_t*)(pb); u.h[1] = *(const v8bf_t*)(pb + 16);
            bfr[nf] = u.v;
        }
        #pragma unroll
        for (int mf = 0; mf < 2; ++mf)
            #pragma unroll
            for (int nf = 0; nf < 4; ++nf) {
                acc[mf][nf] = __builtin_amdgcn_wmma_f32_16x16x32_bf16(
                    false, afh[mf], false, bfr[nf], (short)0, acc[mf][nf], false, false);
                acc[mf][nf] = __builtin_amdgcn_wmma_f32_16x16x32_bf16(
                    false, afl[mf], false, bfr[nf], (short)0, acc[mf][nf], false, false);
            }
        __syncthreads();
    }

    #pragma unroll
    for (int mf = 0; mf < 2; ++mf)
        #pragma unroll
        for (int nf = 0; nf < 4; ++nf) {
            const int gn = bn + waveN * 64 + nf * 16 + l16;
            #pragma unroll
            for (int r = 0; r < 8; ++r) {
                const int gm = bm + waveM * 32 + mf * 16 + half * 8 + r;
                Y[(size_t)gm * DOUT + gn] = acc[mf][nf][r];
            }
        }
}

// ---------------------------------------------------------------------------
extern "C" void kernel_launch(void* const* d_in, const int* in_sizes, int n_in,
                              void* d_out, int out_size, void* d_ws, size_t ws_size,
                              hipStream_t stream) {
    const float* X = (const float*)d_in[0];
    const int*   W = (const int*)d_in[1];
    float*       Y = (float*)d_out;
    const int M = in_sizes[0] / DIN;                 // 8192 tokens

    const size_t nX = (size_t)M * DIN;
    const size_t nW = (size_t)DIN * DOUT;
    const size_t need = nX * 2 * sizeof(unsigned short) + nW * sizeof(unsigned short);

    dim3 grid(DOUT / BN, M / BM);                    // (32, 64)
    if (ws_size >= need) {
        unsigned short* Xhi = (unsigned short*)d_ws;
        unsigned short* Xlo = Xhi + nX;
        unsigned short* Wt  = Xlo + nX;
        cvt_x_kernel<<<dim3((unsigned)(nX / 1024)), dim3(256), 0, stream>>>(X, Xhi, Xlo);
        cvt_w_kernel<<<dim3(DOUT / 64, DIN / 64), dim3(256), 0, stream>>>(W, Wt);
        tern_gemm_wmma_pre<<<grid, dim3(256), 0, stream>>>(Xhi, Xlo, Wt, Y);
    } else {
        tern_gemm_wmma_fallback<<<grid, dim3(256), 0, stream>>>(X, W, Y);
    }
}
